// AttentionBlock_36369783062819
// MI455X (gfx1250) — matrix operand order repaired and checked by reading, not compiled, not run on hardware
//
#include <hip/hip_runtime.h>

#define DEV __device__ __forceinline__

typedef __bf16 bf16;
typedef __attribute__((ext_vector_type(16))) __bf16   v16bf;
typedef __attribute__((ext_vector_type(8)))  float    v8f;
typedef __attribute__((ext_vector_type(4)))  unsigned uint4e;
typedef __attribute__((ext_vector_type(2)))  unsigned uint2e;
typedef __attribute__((ext_vector_type(4)))  float    float4e;
typedef __attribute__((ext_vector_type(2)))  float    float2e;
typedef __attribute__((ext_vector_type(4)))  unsigned u32x4;
typedef __attribute__((ext_vector_type(4)))  int      i32x4;
typedef __attribute__((ext_vector_type(8)))  int      i32x8;

constexpr int   kB     = 16;
constexpr int   kS     = 64 * 64;   // 4096 tokens per batch
constexpr int   kC     = 256;       // channels / units
constexpr int   kN     = kB * kS;   // 65536 rows total
constexpr float kEps   = 1e-3f;
constexpr float kScale = 0.0625f;   // 256^-0.5

union FragU { uint4e u[2]; v16bf v; };
union Pack8 { bf16 h[8]; uint4e u; };
union Pack4 { bf16 h[4]; uint2e u; };

DEV v8f splat8(float x) {
  v8f r;
#pragma unroll
  for (int i = 0; i < 8; ++i) r[i] = x;
  return r;
}

// ---- CDNA5 async global->LDS staging (ASYNCcnt path, no VGPR round-trip) ----
DEV void async_copy_b128(void* lds_dst, const void* gsrc) {
  const unsigned loff = (unsigned)(uintptr_t)lds_dst;
  asm volatile("global_load_async_to_lds_b128 %0, %1, off"
               :: "v"(loff), "v"(gsrc)
               : "memory");
}
DEV void wait_async_all() {
  asm volatile("s_wait_asynccnt 0x0" ::: "memory");
}

// ---- CDNA5 Tensor Data Mover: 1-D contiguous tile, global -> LDS ----
// D# per ISA 08_async_tensor 8.3/8.4: group0 = {count/type, lds_addr, global_addr},
// group1 = {data_size, tensor dims/strides, tile dims}. Issued once per wave,
// tracked with TENSORcnt. nbytes must be a multiple of 4.
DEV void tdm_copy_1d(void* lds_dst, const void* gsrc, unsigned nbytes) {
#if __has_builtin(__builtin_amdgcn_tensor_load_to_lds)
  const unsigned nel = nbytes >> 1;  // bf16 elements, data_size = 2B
  const unsigned long long ga = (unsigned long long)(uintptr_t)gsrc;
  u32x4 g0;
  g0[0] = 1u;                                   // count=1, user descriptor
  g0[1] = (unsigned)(uintptr_t)lds_dst;         // lds_addr (LDS byte offset)
  g0[2] = (unsigned)(ga & 0xffffffffu);         // global_addr[31:0]
  g0[3] = (unsigned)((ga >> 32) & 0x01ffffffu)  // global_addr[56:32]
          | (2u << 30);                         // type = 2 ("image")
  i32x8 g1;
  g1[0] = 0x00010000;            // data_size=1 (2 bytes), no multicast
  g1[1] = (int)(nel << 16);      // tensor_dim0 low bits (bits [79:48])
  g1[2] = 0x00010000;            // tensor_dim1 = 1 (bits [111:80])
  g1[3] = (int)(nel << 16);      // tile_dim0 (bits [127:112])
  g1[4] = 0x00000001;            // tile_dim1 = 1
  g1[5] = (int)nel;              // tensor_dim0_stride low 32 (bits [207:160])
  g1[6] = (int)(nel << 16);      // tensor_dim1_stride low bits (bits [255:208])
  g1[7] = 0;
  i32x4 z4 = {};
#if __clang_major__ >= 23
  i32x8 z8 = {};
  __builtin_amdgcn_tensor_load_to_lds(g0, g1, z4, z4, z8, 0);
#else
  __builtin_amdgcn_tensor_load_to_lds(g0, g1, z4, z4, 0);
#endif
  __builtin_amdgcn_s_wait_tensorcnt(0);
#else
  // fallback: async-to-LDS copies, 16B per lane per issue
  const int lane = threadIdx.x & 31;
  const unsigned chunks = nbytes >> 4;
  for (unsigned c = lane; c < chunks; c += 32)
    async_copy_b128((char*)lds_dst + c * 16, (const char*)gsrc + c * 16);
  wait_async_all();
#endif
}

// A fragment (16x32 MxK) from row-major bf16, row stride ld (elems).
// Per ISA 7.12.2: lanes 0-15 hold K {0..7,16..23}, lanes 16-31 hold K {8..15,24..31}.
DEV v16bf load_a_frag(const bf16* base, int ld) {
  const int lane = threadIdx.x & 31;
  const int row  = lane & 15;
  const int ko   = (lane & 16) ? 8 : 0;
  const bf16* p  = base + row * ld + ko;
  FragU f;
  f.u[0] = *(const uint4e*)(p);
  f.u[1] = *(const uint4e*)(p + 16);
  return f.v;
}

// B fragment (32x16 KxN) from N-major storage bT[n][k], row stride ld (elems).
DEV v16bf load_bT_frag(const bf16* baseT, int ld) {
  const int lane = threadIdx.x & 31;
  const int col  = lane & 15;
  const int ko   = (lane & 16) ? 8 : 0;
  const bf16* p  = baseT + col * ld + ko;
  FragU f;
  f.u[0] = *(const uint4e*)(p);
  f.u[1] = *(const uint4e*)(p + 16);
  return f.v;
}

DEV v8f wmma_bf16(v16bf a, v16bf b, v8f c) {
  return __builtin_amdgcn_wmma_f32_16x16x32_bf16(false, a, false, b,
                                                 (short)0, c, false, false);
}

// Reduce across the 16 lanes of each wave half (C-matrix column lanes).
DEV v8f red16_max(v8f x) {
#pragma unroll
  for (int m = 1; m <= 8; m <<= 1) {
#pragma unroll
    for (int r = 0; r < 8; ++r) x[r] = fmaxf(x[r], __shfl_xor(x[r], m, 32));
  }
  return x;
}
DEV v8f red16_sum(v8f x) {
#pragma unroll
  for (int m = 1; m <= 8; m <<= 1) {
#pragma unroll
    for (int r = 0; r < 8; ++r) x[r] += __shfl_xor(x[r], m, 32);
  }
  return x;
}

// ---------------- GroupNorm ----------------

__global__ __launch_bounds__(256)
void gn_stats_kernel(const float* __restrict__ x, float* __restrict__ stats) {
  const int bg = blockIdx.x;          // 0..127 = b*8+g
  const int b = bg >> 3, g = bg & 7;
  const float* xb = x + (size_t)b * kS * kC + g * 32;
  float s = 0.f, ss = 0.f;
  for (int i = threadIdx.x; i < kS * 32; i += 256) {
    const int sp = i >> 5, c = i & 31;
    const float vv = xb[(size_t)sp * kC + c];
    s += vv;
    ss += vv * vv;
  }
  __shared__ float rs[256], rq[256];
  rs[threadIdx.x] = s;
  rq[threadIdx.x] = ss;
  __syncthreads();
  for (int o = 128; o > 0; o >>= 1) {
    if ((int)threadIdx.x < o) {
      rs[threadIdx.x] += rs[threadIdx.x + o];
      rq[threadIdx.x] += rq[threadIdx.x + o];
    }
    __syncthreads();
  }
  if (threadIdx.x == 0) {
    const float invn = 1.f / (float)(kS * 32);
    const float mean = rs[0] * invn;
    const float var  = rq[0] * invn - mean * mean;
    stats[bg * 2 + 0] = mean;
    stats[bg * 2 + 1] = rsqrtf(var + kEps);
  }
}

__global__ __launch_bounds__(256)
void gn_apply_kernel(const float* __restrict__ x, const float* __restrict__ gamma,
                     const float* __restrict__ beta, const float* __restrict__ stats,
                     float* __restrict__ xn_f, bf16* __restrict__ xn_h) {
  const size_t i = (size_t)blockIdx.x * 256 + threadIdx.x;  // float4 index
  const size_t e = i * 4;
  const int c   = (int)(e & (kC - 1));
  const size_t row = e >> 8;
  const int b   = (int)(row >> 12);
  const int g   = c >> 5;
  const float2e st = ((const float2e*)stats)[b * 8 + g];  // {mean, rstd}
  const float4e xv = ((const float4e*)x)[i];
  const float4e gv = ((const float4e*)gamma)[c >> 2];
  const float4e bv = ((const float4e*)beta)[c >> 2];
  float4e r;
#pragma unroll
  for (int j = 0; j < 4; ++j) r[j] = (xv[j] - st.x) * st.y * gv[j] + bv[j];
  ((float4e*)xn_f)[i] = r;
  Pack4 p;
#pragma unroll
  for (int j = 0; j < 4; ++j) p.h[j] = (bf16)r[j];
  *(uint2e*)(xn_h + e) = p.u;
}

__global__ __launch_bounds__(256)
void cvt_bf16_kernel(const float* __restrict__ a, bf16* __restrict__ o, int n) {
  const int i = (blockIdx.x * 256 + threadIdx.x) * 4;
  if (i >= n) return;
  const float4e xv = *(const float4e*)(a + i);
  Pack4 p;
#pragma unroll
  for (int j = 0; j < 4; ++j) p.h[j] = (bf16)xv[j];
  *(uint2e*)(o + i) = p.u;
}

// ---------------- GEMM: out[M,256] = A[M,256] @ Bw[256,256] + bias (+resid) ----------------
// 256 threads = 8 waves; block tile 128x64, wave tile 32x32 (2x2 WMMA frags).
// A tile staged via global_load_async_to_lds_b128 (ASYNCcnt), B tile transposed via gather.

__global__ __launch_bounds__(256)
void gemm256_kernel(const bf16* __restrict__ A, const bf16* __restrict__ Bw,
                    const float* __restrict__ bias, const float* __restrict__ resid,
                    bf16* __restrict__ outh, float* __restrict__ outf) {
  __shared__ bf16 As[128][32];
  __shared__ bf16 Bts[64][32];  // transposed: Bts[n][k]
  const int tid  = threadIdx.x;
  const int lane = tid & 31, wave = tid >> 5;
  const int mBlk = blockIdx.y * 128;
  const int nBlk = blockIdx.x * 64;
  const int wm   = (wave >> 1) * 32;
  const int wn   = (wave & 1) * 32;
  v8f acc[2][2] = {};

  for (int k0 = 0; k0 < kC; k0 += 32) {
    {  // A tile: 128x32, async straight into LDS (2 x b128 per thread)
      const int row = tid >> 1, half = (tid & 1) * 16;
      const bf16* src = A + (size_t)(mBlk + row) * kC + k0 + half;
      async_copy_b128(&As[row][half], src);
      async_copy_b128(&As[row][half + 8], src + 8);
    }
    {  // B tile transposed: each thread gathers an 8-deep K column
      const int n = tid & 63, kq = (tid >> 6) * 8;
      Pack8 t;
#pragma unroll
      for (int j = 0; j < 8; ++j)
        t.h[j] = Bw[(size_t)(k0 + kq + j) * kC + nBlk + n];
      *(uint4e*)(&Bts[n][kq]) = t.u;
    }
    wait_async_all();
    __syncthreads();
    const v16bf a0 = load_a_frag(&As[wm][0], 32);
    const v16bf a1 = load_a_frag(&As[wm + 16][0], 32);
    const v16bf b0 = load_bT_frag(&Bts[wn][0], 32);
    const v16bf b1 = load_bT_frag(&Bts[wn + 16][0], 32);
    acc[0][0] = wmma_bf16(a0, b0, acc[0][0]);
    acc[0][1] = wmma_bf16(a0, b1, acc[0][1]);
    acc[1][0] = wmma_bf16(a1, b0, acc[1][0]);
    acc[1][1] = wmma_bf16(a1, b1, acc[1][1]);
    __syncthreads();
  }

  const int col = lane & 15, rowb = (lane >> 4) * 8;  // C/D layout (ISA 7.12.2)
#pragma unroll
  for (int i = 0; i < 2; ++i)
#pragma unroll
    for (int j = 0; j < 2; ++j) {
      const int gr0 = mBlk + wm + i * 16 + rowb;
      const int gc  = nBlk + wn + j * 16 + col;
      const float bb = bias[gc];
#pragma unroll
      for (int r = 0; r < 8; ++r) {
        const size_t idx = (size_t)(gr0 + r) * kC + gc;
        float vv = acc[i][j][r] + bb;
        if (resid) vv += resid[idx];
        if (outf) outf[idx] = vv;
        if (outh) outh[idx] = (bf16)vv;
      }
    }
}

// ---------------- Flash attention ----------------
// Block: 128 threads = 4 waves; 64 query rows per block (16 per wave).
// Q tile staged by the Tensor Data Mover (one 32 KB contiguous D# copy, TENSORcnt),
// streams 32-key blocks, online softmax, never materializes the 4096x4096 scores.

__global__ __launch_bounds__(128)
void attention_kernel(const bf16* __restrict__ q, const bf16* __restrict__ k,
                      const bf16* __restrict__ v, bf16* __restrict__ o) {
  __shared__ bf16 Qs[64][256];       // 32 KB
  __shared__ bf16 Vt[256][32];       // 16 KB, Vt[c][key]
  __shared__ bf16 Ps[4][16][32];     // 4 KB, per-wave P scratch
  const int tid  = threadIdx.x;
  const int lane = tid & 31, wave = tid >> 5;
  const int b    = blockIdx.y;
  const int qBlk = blockIdx.x * 64;
  const size_t base = (size_t)b * kS * kC;

  // stage Q tile once: rows are adjacent in memory -> one contiguous 32 KB TDM copy
  if (wave == 0)
    tdm_copy_1d(&Qs[0][0], q + base + (size_t)qBlk * kC, 64 * kC * 2);
  __syncthreads();

  const int wrow = wave * 16;
  v8f O[16] = {};
  v8f mrow = splat8(-1e30f);
  v8f lrow = {};

  for (int kb = 0; kb < kS; kb += 32) {
    // stage V^T for this key block: thread owns 2 channels, walks 32 keys
#pragma unroll
    for (int cc = 0; cc < 2; ++cc) {
      const int c = tid + cc * 128;
      const bf16* vp = v + base + (size_t)kb * kC + c;
#pragma unroll
      for (int kk = 0; kk < 32; kk += 8) {
        Pack8 t;
#pragma unroll
        for (int j = 0; j < 8; ++j) t.h[j] = vp[(size_t)(kk + j) * kC];
        *(uint4e*)(&Vt[c][kk]) = t.u;
      }
    }
    __syncthreads();

    // prefetch next key block of K and V into cache (global_prefetch_b8)
    if (kb + 32 < kS) {
      __builtin_prefetch(k + base + (size_t)(kb + 32) * kC + tid * 64, 0, 1);
      __builtin_prefetch(v + base + (size_t)(kb + 32) * kC + tid * 64, 0, 1);
    }

    // S = (Q Kt) * scale : B frags come straight from global k
    v8f s0 = {}, s1 = {};
    const bf16* kp = k + base + (size_t)kb * kC;
#pragma unroll
    for (int kc = 0; kc < kC; kc += 32) {
      const v16bf a  = load_a_frag(&Qs[wrow][kc], kC);
      const v16bf b0 = load_bT_frag(kp + kc, kC);             // keys kb..kb+15
      const v16bf b1 = load_bT_frag(kp + 16 * kC + kc, kC);   // keys kb+16..kb+31
      s0 = wmma_bf16(a, b0, s0);
      s1 = wmma_bf16(a, b1, s1);
    }
#pragma unroll
    for (int r = 0; r < 8; ++r) { s0[r] *= kScale; s1[r] *= kScale; }

    // online softmax update (row stats replicated across the 16 column lanes)
    v8f rm;
#pragma unroll
    for (int r = 0; r < 8; ++r) rm[r] = fmaxf(s0[r], s1[r]);
    rm = red16_max(rm);
    v8f mnew, alpha, p0, p1;
#pragma unroll
    for (int r = 0; r < 8; ++r) mnew[r] = fmaxf(mrow[r], rm[r]);
#pragma unroll
    for (int r = 0; r < 8; ++r) {
      alpha[r] = __expf(mrow[r] - mnew[r]);
      p0[r]    = __expf(s0[r] - mnew[r]);
      p1[r]    = __expf(s1[r] - mnew[r]);
    }
    const v8f rsum = red16_sum(p0 + p1);
#pragma unroll
    for (int r = 0; r < 8; ++r) lrow[r] = lrow[r] * alpha[r] + rsum[r];
    mrow = mnew;

    {  // C-layout -> A-layout via wave-private LDS scratch (in-order per wave)
      const int col = lane & 15, rowb = (lane >> 4) * 8;
#pragma unroll
      for (int r = 0; r < 8; ++r) {
        Ps[wave][rowb + r][col]      = (bf16)p0[r];
        Ps[wave][rowb + r][16 + col] = (bf16)p1[r];
      }
    }
#pragma unroll
    for (int f = 0; f < 16; ++f) O[f] = O[f] * alpha;

    // O += P @ V
    const v16bf pa = load_a_frag(&Ps[wave][0][0], 32);
#pragma unroll
    for (int nf = 0; nf < 16; ++nf) {
      const v16bf bfr = load_bT_frag(&Vt[nf * 16][0], 32);
      O[nf] = wmma_bf16(pa, bfr, O[nf]);
    }
    __syncthreads();  // protect Vt before next key block overwrites it
  }

  // finalize: O / l -> bf16 output
  v8f inv;
#pragma unroll
  for (int r = 0; r < 8; ++r) inv[r] = 1.f / lrow[r];
  const int col = lane & 15, rowb = (lane >> 4) * 8;
#pragma unroll
  for (int nf = 0; nf < 16; ++nf)
#pragma unroll
    for (int r = 0; r < 8; ++r)
      o[base + (size_t)(qBlk + wrow + rowb + r) * kC + nf * 16 + col] =
          (bf16)(O[nf][r] * inv[r]);
}

// ---------------- launch ----------------

extern "C" void kernel_launch(void* const* d_in, const int* in_sizes, int n_in,
                              void* d_out, int out_size, void* d_ws, size_t ws_size,
                              hipStream_t stream) {
  (void)in_sizes; (void)n_in; (void)out_size; (void)ws_size;
  const float* x     = (const float*)d_in[0];
  const float* gamma = (const float*)d_in[1];
  const float* beta  = (const float*)d_in[2];
  const float* Wq    = (const float*)d_in[3];
  const float* bq    = (const float*)d_in[4];
  const float* Wk    = (const float*)d_in[5];
  const float* bk    = (const float*)d_in[6];
  const float* Wv    = (const float*)d_in[7];
  const float* bv    = (const float*)d_in[8];
  const float* Wp    = (const float*)d_in[9];
  const float* bp    = (const float*)d_in[10];
  float* out = (float*)d_out;

  char* ws = (char*)d_ws;
  size_t off = 0;
  auto take = [&](size_t bytes) -> char* {
    char* p = ws + off;
    off = (off + bytes + 255) & ~(size_t)255;
    return p;
  };
  float* xn_f  = (float*)take((size_t)kN * kC * 4);
  bf16*  xn_h  = (bf16*) take((size_t)kN * kC * 2);
  bf16*  qh    = (bf16*) take((size_t)kN * kC * 2);
  bf16*  kh    = (bf16*) take((size_t)kN * kC * 2);
  bf16*  vh    = (bf16*) take((size_t)kN * kC * 2);
  bf16*  ah    = (bf16*) take((size_t)kN * kC * 2);
  bf16*  Wqh   = (bf16*) take((size_t)kC * kC * 2);
  bf16*  Wkh   = (bf16*) take((size_t)kC * kC * 2);
  bf16*  Wvh   = (bf16*) take((size_t)kC * kC * 2);
  bf16*  Wph   = (bf16*) take((size_t)kC * kC * 2);
  float* stats = (float*)take((size_t)kB * 8 * 2 * 4);

  gn_stats_kernel<<<dim3(kB * 8), dim3(256), 0, stream>>>(x, stats);
  gn_apply_kernel<<<dim3((kN * kC) / 1024), dim3(256), 0, stream>>>(
      x, gamma, beta, stats, xn_f, xn_h);

  cvt_bf16_kernel<<<dim3(64), dim3(256), 0, stream>>>(Wq, Wqh, kC * kC);
  cvt_bf16_kernel<<<dim3(64), dim3(256), 0, stream>>>(Wk, Wkh, kC * kC);
  cvt_bf16_kernel<<<dim3(64), dim3(256), 0, stream>>>(Wv, Wvh, kC * kC);
  cvt_bf16_kernel<<<dim3(64), dim3(256), 0, stream>>>(Wp, Wph, kC * kC);

  dim3 ggrid(kC / 64, kN / 128);
  gemm256_kernel<<<ggrid, 256, 0, stream>>>(xn_h, Wqh, bq, nullptr, qh, nullptr);
  gemm256_kernel<<<ggrid, 256, 0, stream>>>(xn_h, Wkh, bk, nullptr, kh, nullptr);
  gemm256_kernel<<<ggrid, 256, 0, stream>>>(xn_h, Wvh, bv, nullptr, vh, nullptr);

  attention_kernel<<<dim3(kS / 64, kB), dim3(128), 0, stream>>>(qh, kh, vh, ah);

  gemm256_kernel<<<ggrid, 256, 0, stream>>>(ah, Wph, bp, xn_f, nullptr, out);
}
